// SoftDPPCausalSelfAttention_6236292514011
// MI455X (gfx1250) — matrix operand order repaired and checked by reading, not compiled, not run on hardware
//
#include <hip/hip_runtime.h>
#include <hip/hip_bf16.h>
#include <math.h>

typedef __bf16 bf16_t;
typedef __attribute__((ext_vector_type(16))) __bf16 v16bf;
typedef __attribute__((ext_vector_type(8)))  __bf16 v8bf;
typedef __attribute__((ext_vector_type(8)))  float  v8f;

#define WMMA_BF16(a, b, c) \
  __builtin_amdgcn_wmma_f32_16x16x32_bf16(false, (a), false, (b), (short)0, (c), false, false)

// ---- CDNA5 async global->LDS copy (ASYNCcnt-tracked) ----------------------
// LDS generic addresses carry the LDS byte offset in bits [31:0] (ISA 10.2),
// so truncating a generic pointer to __shared__ yields the vdst address.
__device__ __forceinline__ void async_copy_b128(void* lds_ptr, const void* gptr) {
  uint32_t lds = (uint32_t)(uintptr_t)lds_ptr;
  asm volatile("global_load_async_to_lds_b128 %0, %1, off"
               :: "v"(lds), "v"(gptr) : "memory");
}
__device__ __forceinline__ void wait_async0() {
  asm volatile("s_wait_asynccnt 0x0" ::: "memory");
}

// ---- WMMA fragment loaders (bf16, 16x16x32 shape) -------------------------
// A (16x32, MxK): lane l holds row (base + (l&15)); elements 0..7 at
// K = kk + (l>>4)*8 + {0..7}, elements 8..15 at +16. Two 16B loads.
__device__ __forceinline__ v16bf load_a_frag(const bf16_t* __restrict__ base,
                                             int row, int stride, int kk, int lane) {
  const bf16_t* p = base + (size_t)row * stride + kk + ((lane >> 4) << 3);
  union { v16bf v; v8bf h[2]; } u;
  u.h[0] = *(const v8bf*)p;
  u.h[1] = *(const v8bf*)(p + 16);
  return u.v;
}
// B (32x16, KxN): lane l holds column (base + (l&15)); elements 0..7 at
// K = kk + (l>>4)*8 + {0..7}, elements 8..15 at +16. `stride` = per-column stride,
// i.e. B is sourced from row-major storage of B^T.
__device__ __forceinline__ v16bf load_b_frag(const bf16_t* __restrict__ base,
                                             int col, int stride, int kk, int lane) {
  const bf16_t* p = base + (size_t)col * stride + kk + ((lane >> 4) << 3);
  union { v16bf v; v8bf h[2]; } u;
  u.h[0] = *(const v8bf*)p;
  u.h[1] = *(const v8bf*)(p + 16);
  return u.v;
}

// ---- Register-pipelined 16x64 GEMM strip (per wave) -----------------------
template <int KTOT>
__device__ __forceinline__ void gemm16x64(const bf16_t* __restrict__ Abase, int arow, int astride,
                                          const bf16_t* __restrict__ Bbase, int bcol, int bstride,
                                          int lane, v8f acc[4]) {
  v16bf a  = load_a_frag(Abase, arow, astride, 0, lane);
  v16bf b0 = load_b_frag(Bbase, bcol,      bstride, 0, lane);
  v16bf b1 = load_b_frag(Bbase, bcol + 16, bstride, 0, lane);
  v16bf b2 = load_b_frag(Bbase, bcol + 32, bstride, 0, lane);
  v16bf b3 = load_b_frag(Bbase, bcol + 48, bstride, 0, lane);
#pragma unroll
  for (int kk = 0; kk < KTOT; kk += 32) {
    v16bf ac = a, c0 = b0, c1 = b1, c2 = b2, c3 = b3;
    if (kk + 32 < KTOT) {  // compile-time resolved (full unroll)
      a  = load_a_frag(Abase, arow, astride, kk + 32, lane);
      b0 = load_b_frag(Bbase, bcol,      bstride, kk + 32, lane);
      b1 = load_b_frag(Bbase, bcol + 16, bstride, kk + 32, lane);
      b2 = load_b_frag(Bbase, bcol + 32, bstride, kk + 32, lane);
      b3 = load_b_frag(Bbase, bcol + 48, bstride, kk + 32, lane);
    }
    acc[0] = WMMA_BF16(ac, c0, acc[0]);
    acc[1] = WMMA_BF16(ac, c1, acc[1]);
    acc[2] = WMMA_BF16(ac, c2, acc[2]);
    acc[3] = WMMA_BF16(ac, c3, acc[3]);
  }
}

// ---- Block-cooperative 128x64 GEMM with async-LDS-staged B ---------------
// 8 waves; wave w owns rows [aRow0 + 16w, +16). B's 64x32 k-chunk is staged
// into double-buffered LDS (rows padded to 40 elems) via one
// global_load_async_to_lds_b128 per thread, overlapping the 4-WMMA consume
// phase of the other buffer. Requires blockDim.x == 256.
template <int KTOT>
__device__ __forceinline__ void gemm_blk128x64(const bf16_t* __restrict__ Abase, int aRow0,
                                               int astride,
                                               const bf16_t* __restrict__ Bbase, int bCol0,
                                               int bstride, bf16_t* Bs /*[2][64][40]*/,
                                               v8f acc[4]) {
  const int tid = threadIdx.x, lane = tid & 31, wid = tid >> 5;
  const int arow = aRow0 + (wid << 4) + (lane & 15);
  const int sc = tid >> 2, sseg = tid & 3;  // this thread's staged row/segment
  const bf16_t* gsrc = Bbase + (size_t)(bCol0 + sc) * bstride + (sseg << 3);
  bf16_t* ldst = Bs + (size_t)sc * 40 + (sseg << 3);

  async_copy_b128(ldst, gsrc);                       // stage buf0, kk=0
  v16bf a = load_a_frag(Abase, arow, astride, 0, lane);
  wait_async0();
  __syncthreads();

  constexpr int NSTEP = KTOT / 32;
#pragma unroll
  for (int s = 0; s < NSTEP; ++s) {
    const int cur = s & 1;
    if (s + 1 < NSTEP)                                // stage next chunk (async)
      async_copy_b128(ldst + (cur ^ 1) * (64 * 40), gsrc + (s + 1) * 32);
    v16bf ac = a;
    if (s + 1 < NSTEP)
      a = load_a_frag(Abase, arow, astride, (s + 1) * 32, lane);
    const bf16_t* Bl = Bs + cur * (64 * 40);
    acc[0] = WMMA_BF16(ac, load_b_frag(Bl, (lane & 15),      40, 0, lane), acc[0]);
    acc[1] = WMMA_BF16(ac, load_b_frag(Bl, (lane & 15) + 16, 40, 0, lane), acc[1]);
    acc[2] = WMMA_BF16(ac, load_b_frag(Bl, (lane & 15) + 32, 40, 0, lane), acc[2]);
    acc[3] = WMMA_BF16(ac, load_b_frag(Bl, (lane & 15) + 48, 40, 0, lane), acc[3]);
    if (s + 1 < NSTEP) wait_async0();                 // next buffer landed
    __syncthreads();                                  // all reads of cur done
  }
}

// ---- K0: f32 -> bf16 cast -------------------------------------------------
__global__ void k_cast_bf16(const float* __restrict__ in, bf16_t* __restrict__ out, int n) {
  int i = blockIdx.x * blockDim.x + threadIdx.x;
  if (i < n) out[i] = (bf16_t)in[i];
}

// ---- K1: qkv = x @ W_attn^T + b; scatter into head layouts ---------------
// M=2048 (B*T), N=1536, K=512. Block computes 128x64. Writes:
//   qb [BH][T][hs], kb [BH][T][hs], ktb [BH][hs][T], vtb [BH][hs][T]  (bf16)
__global__ void __launch_bounds__(256) k_qkv(const bf16_t* __restrict__ xb,
                                             const bf16_t* __restrict__ wab,
                                             const float* __restrict__ b_attn,
                                             bf16_t* __restrict__ qb, bf16_t* __restrict__ kb,
                                             bf16_t* __restrict__ ktb, bf16_t* __restrict__ vtb) {
  __shared__ alignas(16) bf16_t Bs[2 * 64 * 40];
  int lane = threadIdx.x & 31, wid = threadIdx.x >> 5;
  int aRow0 = (blockIdx.x / 24) << 7;
  int tileN = (blockIdx.x % 24) << 6;
  v8f acc[4] = {};
  gemm_blk128x64<512>(xb, aRow0, 512, wab, tileN, 512, Bs, acc);
  int tileM = aRow0 + (wid << 4);
  int rhalf = (lane >> 4) << 3;
#pragma unroll
  for (int j = 0; j < 4; ++j) {
    int n = tileN + (j << 4) + (lane & 15);
    float bias = b_attn[n];
    int p = n >> 9, c = n & 511, hh = c >> 6, d = c & 63;
#pragma unroll
    for (int r = 0; r < 8; ++r) {
      int m = tileM + r + rhalf;
      int bb = m >> 8, t = m & 255, bh = bb * 8 + hh;
      bf16_t bv = (bf16_t)(acc[j][r] + bias);
      if (p == 0) {
        qb[((size_t)bh * 256 + t) * 64 + d] = bv;
      } else if (p == 1) {
        kb[((size_t)bh * 256 + t) * 64 + d] = bv;
        ktb[((size_t)bh * 64 + d) * 256 + t] = bv;
      } else {
        vtb[((size_t)bh * 64 + d) * 256 + t] = bv;
      }
    }
  }
}

// ---- K2: scores = q k^T * 0.125, causal mask, softmax -> attb bf16 -------
// One block = one (bh, 16-row chunk). 8 waves, each computes two 16x16 tiles.
__global__ void k_attn(const bf16_t* __restrict__ qb, const bf16_t* __restrict__ kb,
                       bf16_t* __restrict__ attb) {
  __shared__ float S[16][256];
  __shared__ float red[16][16];
  __shared__ float rowstat[16];
  int tid = threadIdx.x, lane = tid & 31, wid = tid >> 5;
  int bh = blockIdx.x >> 4;
  int rowTile = (blockIdx.x & 15) << 4;
  const bf16_t* qbh = qb + (size_t)bh * 256 * 64;
  const bf16_t* kbh = kb + (size_t)bh * 256 * 64;
  int arow = rowTile + (lane & 15);
  int bcol0 = ((wid * 2) << 4) + (lane & 15);
  int bcol1 = bcol0 + 16;
  v16bf a0  = load_a_frag(qbh, arow, 64, 0, lane);
  v16bf a1  = load_a_frag(qbh, arow, 64, 32, lane);
  v16bf b00 = load_b_frag(kbh, bcol0, 64, 0, lane);
  v16bf b01 = load_b_frag(kbh, bcol0, 64, 32, lane);
  v16bf b10 = load_b_frag(kbh, bcol1, 64, 0, lane);
  v16bf b11 = load_b_frag(kbh, bcol1, 64, 32, lane);
  v8f acc0 = {}, acc1 = {};
  acc0 = WMMA_BF16(a0, b00, acc0);
  acc1 = WMMA_BF16(a0, b10, acc1);
  acc0 = WMMA_BF16(a1, b01, acc0);
  acc1 = WMMA_BF16(a1, b11, acc1);
  int rhalf = (lane >> 4) << 3;
#pragma unroll
  for (int r = 0; r < 8; ++r) {
    int mloc = r + rhalf;
    int row = rowTile + mloc;
    float v0 = acc0[r] * 0.125f;               // 1/sqrt(64)
    float v1 = acc1[r] * 0.125f;
    if (bcol0 > row) v0 = -INFINITY;           // causal
    if (bcol1 > row) v1 = -INFINITY;
    S[mloc][bcol0] = v0;
    S[mloc][bcol1] = v1;
  }
  __syncthreads();
  int row = tid >> 4, seg = tid & 15;
  float* Sr = &S[row][seg << 4];
  float mx = Sr[0];
#pragma unroll
  for (int i = 1; i < 16; ++i) mx = fmaxf(mx, Sr[i]);
  red[row][seg] = mx;
  __syncthreads();
  if (seg == 0) {
    float m2 = red[row][0];
    for (int i = 1; i < 16; ++i) m2 = fmaxf(m2, red[row][i]);
    rowstat[row] = m2;
  }
  __syncthreads();
  float rm = rowstat[row];
  float sum = 0.f;
#pragma unroll
  for (int i = 0; i < 16; ++i) { float e = __expf(Sr[i] - rm); Sr[i] = e; sum += e; }
  red[row][seg] = sum;
  __syncthreads();
  if (seg == 0) {
    float s2 = 0.f;
    for (int i = 0; i < 16; ++i) s2 += red[row][i];
    rowstat[row] = s2;
  }
  __syncthreads();
  float inv = 1.0f / rowstat[row];
  bf16_t* arow_out = attb + ((size_t)bh * 256 + rowTile + row) * 256 + (seg << 4);
#pragma unroll
  for (int i = 0; i < 16; ++i) arow_out[i] = (bf16_t)(Sr[i] * inv);
}

// ---- K3: y_bh = att @ v  -> ycat [B][T][C] bf16 ---------------------------
__global__ void k_av(const bf16_t* __restrict__ attb, const bf16_t* __restrict__ vtb,
                     bf16_t* __restrict__ ycat) {
  int lane  = threadIdx.x & 31;
  int waveG = blockIdx.x * (blockDim.x >> 5) + (threadIdx.x >> 5);
  int bh = waveG >> 4;
  int tileM = (waveG & 15) << 4;
  const bf16_t* abh = attb + (size_t)bh * 256 * 256;
  const bf16_t* vbh = vtb + (size_t)bh * 64 * 256;
  v8f acc[4] = {};
  gemm16x64<256>(abh, tileM + (lane & 15), 256, vbh, (lane & 15), 256, lane, acc);
  int rhalf = (lane >> 4) << 3;
  int bb = bh >> 3, hh = bh & 7;
#pragma unroll
  for (int j = 0; j < 4; ++j) {
    int d = (j << 4) + (lane & 15);
#pragma unroll
    for (int r = 0; r < 8; ++r) {
      int t = tileM + r + rhalf;
      ycat[((size_t)bb * 256 + t) * 512 + hh * 64 + d] = (bf16_t)acc[j][r];
    }
  }
}

// ---- K4: out = ycat @ W_proj^T + b_proj (f32 output) ----------------------
__global__ void __launch_bounds__(256) k_proj(const bf16_t* __restrict__ ycat,
                                              const bf16_t* __restrict__ wpb,
                                              const float* __restrict__ b_proj,
                                              float* __restrict__ out) {
  __shared__ alignas(16) bf16_t Bs[2 * 64 * 40];
  int lane = threadIdx.x & 31, wid = threadIdx.x >> 5;
  int aRow0 = (blockIdx.x >> 3) << 7;
  int tileN = (blockIdx.x & 7) << 6;
  v8f acc[4] = {};
  gemm_blk128x64<512>(ycat, aRow0, 512, wpb, tileN, 512, Bs, acc);
  int tileM = aRow0 + (wid << 4);
  int rhalf = (lane >> 4) << 3;
#pragma unroll
  for (int j = 0; j < 4; ++j) {
    int n = tileN + (j << 4) + (lane & 15);
    float bias = b_proj[n];
#pragma unroll
    for (int r = 0; r < 8; ++r) {
      int m = tileM + r + rhalf;
      out[(size_t)m * 512 + n] = acc[j][r] + bias;
    }
  }
}

// ---- K5: per-token DPP term via determinant lemma -------------------------
// G = diag(s) K K^T diag(s) + 1e-6 I is rank-64 + eps*I, so
// det(G) = (1e-6)^192 * det(1e-6 I_64 + M^T M), M = diag(s) K  (256x64).
// A64 = M^T M + 1e-6 I via WMMA from LDS; Cholesky logdet; det = exp(...)
// underflows f32 exactly like jnp.linalg.det; contrib = log(det + 1e-8).
__global__ void __launch_bounds__(128) k_logdet(const bf16_t* __restrict__ attb,
                                                const bf16_t* __restrict__ ktb,
                                                float* __restrict__ contrib) {
  __shared__ bf16_t Mt[64][264];   // padded rows to dodge bank conflicts
  __shared__ float sArr[256];
  __shared__ float A64[64][65];
  __shared__ float rbuf[128];
  int tid = threadIdx.x, lane = tid & 31, wid = tid >> 5;
  int bh = blockIdx.x >> 8;
  int tq = blockIdx.x & 255;
  const bf16_t* wrow = attb + ((size_t)bh * 256 + tq) * 256;
  for (int t = tid; t < 256; t += 128) sArr[t] = sqrtf((float)wrow[t] + 1e-8f);
  __syncthreads();
  const bf16_t* kt = ktb + (size_t)bh * 64 * 256;
  for (int idx = tid; idx < 64 * 256; idx += 128) {
    int d = idx >> 8, t = idx & 255;
    Mt[d][t] = (bf16_t)((float)kt[idx] * sArr[t]);
  }
  __syncthreads();
  // wave `wid` computes 16-row tile i=wid of D = Mt * Mt^T (64x64)
  v8f acc[4] = {};
  gemm16x64<256>(&Mt[0][0], (wid << 4) + (lane & 15), 264,
                 &Mt[0][0], (lane & 15), 264, lane, acc);
  int rhalf = (lane >> 4) << 3;
#pragma unroll
  for (int j = 0; j < 4; ++j) {
    int n = (j << 4) + (lane & 15);
#pragma unroll
    for (int r = 0; r < 8; ++r) {
      int m = (wid << 4) + r + rhalf;
      A64[m][n] = acc[j][r] + ((m == n) ? 1e-6f : 0.0f);
    }
  }
  __syncthreads();
  // Cooperative right-looking Cholesky (SPD by construction)
  for (int j = 0; j < 64; ++j) {
    if (tid == 0) A64[j][j] = sqrtf(A64[j][j]);
    __syncthreads();
    if (tid < 63 - j) { int i = j + 1 + tid; A64[i][j] /= A64[j][j]; }
    __syncthreads();
    if (tid < 63 - j) {
      int i = j + 1 + tid;
      float Lij = A64[i][j];
      for (int k2 = j + 1; k2 <= i; ++k2) A64[i][k2] -= Lij * A64[k2][j];
    }
    __syncthreads();
  }
  rbuf[tid] = (tid < 64) ? __logf(A64[tid][tid]) : 0.0f;
  __syncthreads();
  for (int s = 64; s > 0; s >>= 1) {
    if (tid < s) rbuf[tid] += rbuf[tid + s];
    __syncthreads();
  }
  if (tid == 0) {
    float logdet = 2.0f * rbuf[0] + 192.0f * __logf(1e-6f);
    float det = __expf(logdet);  // underflows to 0 in f32 exactly like jnp det
    contrib[blockIdx.x] = __logf(det + 1e-8f);
  }
}

// ---- K6: deterministic fixed-order reduction -> penalty -------------------
__global__ void k_finalize(const float* __restrict__ contrib, float* __restrict__ out,
                           int n, int outIdx) {
  __shared__ float buf[256];
  float s = 0.f;
  for (int i = threadIdx.x; i < n; i += 256) s += contrib[i];
  buf[threadIdx.x] = s;
  __syncthreads();
  for (int st = 128; st > 0; st >>= 1) {
    if (threadIdx.x < st) buf[threadIdx.x] += buf[threadIdx.x + st];
    __syncthreads();
  }
  if (threadIdx.x == 0) out[outIdx] = -0.01f * buf[0];
}

extern "C" void kernel_launch(void* const* d_in, const int* in_sizes, int n_in,
                              void* d_out, int out_size, void* d_ws, size_t ws_size,
                              hipStream_t stream) {
  const float* x      = (const float*)d_in[0];
  const float* W_attn = (const float*)d_in[1];
  const float* b_attn = (const float*)d_in[2];
  const float* W_proj = (const float*)d_in[3];
  const float* b_proj = (const float*)d_in[4];
  float* out = (float*)d_out;

  char* ws = (char*)d_ws;
  size_t off = 0;
  auto take = [&](size_t bytes) {
    char* p = ws + off;
    off = (off + bytes + 255) & ~(size_t)255;
    return p;
  };
  bf16_t* xb   = (bf16_t*)take((size_t)2048 * 512 * 2);
  bf16_t* wab  = (bf16_t*)take((size_t)1536 * 512 * 2);
  bf16_t* wpb  = (bf16_t*)take((size_t)512 * 512 * 2);
  bf16_t* qb   = (bf16_t*)take((size_t)64 * 256 * 64 * 2);
  bf16_t* kb   = (bf16_t*)take((size_t)64 * 256 * 64 * 2);
  bf16_t* ktb  = (bf16_t*)take((size_t)64 * 64 * 256 * 2);
  bf16_t* vtb  = (bf16_t*)take((size_t)64 * 64 * 256 * 2);
  bf16_t* attb = (bf16_t*)take((size_t)64 * 256 * 256 * 2);
  bf16_t* ycat = (bf16_t*)take((size_t)2048 * 512 * 2);
  float*  contrib = (float*)take((size_t)16384 * 4);

  k_cast_bf16<<<(2048 * 512 + 255) / 256, 256, 0, stream>>>(x, xb, 2048 * 512);
  k_cast_bf16<<<(1536 * 512 + 255) / 256, 256, 0, stream>>>(W_attn, wab, 1536 * 512);
  k_cast_bf16<<<(512 * 512 + 255) / 256, 256, 0, stream>>>(W_proj, wpb, 512 * 512);
  k_qkv<<<384, 256, 0, stream>>>(xb, wab, b_attn, qb, kb, ktb, vtb);
  k_attn<<<1024, 256, 0, stream>>>(qb, kb, attb);
  k_av<<<128, 256, 0, stream>>>(attb, vtb, ycat);
  k_proj<<<128, 256, 0, stream>>>(ycat, wpb, b_proj, out);
  k_logdet<<<16384, 128, 0, stream>>>(attb, ktb, contrib);
  k_finalize<<<1, 256, 0, stream>>>(contrib, out, 16384, out_size - 1);
}
